// GraphConvolutionalNetwork_26087631356420
// MI455X (gfx1250) — matrix operand order repaired and checked by reading, not compiled, not run on hardware
//
#include <hip/hip_runtime.h>
#include <hip/hip_fp16.h>

#define N_NODE   100000
#define N_EDGE   1600000
#define IN_DIM   128
#define HID_DIM  64
#define OUT_DIM  32
#define NUM_GRAPHS 256
#define N_ROW_TILES (N_NODE / 16)   // 6250, exact

typedef __attribute__((ext_vector_type(16))) _Float16 v16h;
typedef __attribute__((ext_vector_type(8)))  _Float16 v8h;
typedef __attribute__((ext_vector_type(8)))  float    v8f;

// ---------------------------------------------------------------- utilities

__global__ void zero_f32_kernel(float* __restrict__ p, int n) {
    int i = blockIdx.x * blockDim.x + threadIdx.x;
    if (i < n) p[i] = 0.0f;
}

__global__ void f32_to_f16_kernel(const float* __restrict__ x,
                                  _Float16* __restrict__ y, int n) {
    int i = blockIdx.x * blockDim.x + threadIdx.x;
    if (i < n) y[i] = (_Float16)x[i];
}

// ------------------------------------------------------------------ degrees

__global__ void degree_accum_kernel(const int* __restrict__ snd,
                                    const int* __restrict__ rcv,
                                    float* __restrict__ deg_s,
                                    float* __restrict__ deg_r, int ne) {
    int e = blockIdx.x * blockDim.x + threadIdx.x;
    if (e < ne) {
        atomicAdd(&deg_s[snd[e]], 1.0f);
        atomicAdd(&deg_r[rcv[e]], 1.0f);
    }
}

__global__ void deg_to_invsqrt_kernel(float* __restrict__ d, int n) {
    int i = blockIdx.x * blockDim.x + threadIdx.x;
    if (i < n) d[i] = rsqrtf(fmaxf(d[i], 1.0f));
}

// ------------------------------------------------- WMMA GEMM  H = (X@W + b)*s
// One wave32 per 16x16 output tile. K advances 32 per v_wmma_f32_16x16x32_f16.
// A layout (16-bit A 16x32, ISA 7.12.2): lanes 0-15 (M=lane): elems0-7 = K
// kb+0..7, elems8-15 = K kb+16..23; lanes 16-31 (M=lane-16): K kb+8..15 /
// kb+24..31.  B layout (KxN): lane&15 = N column, 16 halves = K in the same order as A.
// C/D layout: N = lane&15, M = vgpr + 8*(lane>=16).

template <int DIN, int DOUT>
__global__ __launch_bounds__(32)
void gemm_wmma_kernel(const _Float16* __restrict__ Xh,
                      const _Float16* __restrict__ Wh,
                      const float* __restrict__ bias,
                      const float* __restrict__ inv_s,
                      float* __restrict__ H) {
    const int rowTile = blockIdx.x;            // 0..N_ROW_TILES-1
    const int colTile = blockIdx.y;            // 0..DOUT/16-1
    const int lane    = threadIdx.x;           // 0..31
    const int hi      = lane >> 4;             // half-wave select
    const int lm      = lane & 15;
    const int row0    = rowTile * 16;
    const int n0      = colTile * 16;
    const int n       = n0 + lm;
    const int koff    = hi ? 8 : 0;

    v8f acc = {};
    const _Float16* arow = Xh + (size_t)(row0 + lm) * DIN + koff;
#pragma unroll
    for (int kb = 0; kb < DIN; kb += 32) {
        v8h alo = *(const v8h*)(arow + kb);        // K kb+koff .. +7
        v8h ahi = *(const v8h*)(arow + kb + 16);   // K kb+koff+16 .. +23
        v16h a;
#pragma unroll
        for (int j = 0; j < 8; ++j) { a[j] = alo[j]; a[j + 8] = ahi[j]; }
        // B: this lane holds column n, K kb+koff..+7 / kb+koff+16..+23
        v16h b;
#pragma unroll
        for (int j = 0; j < 8; ++j) {
            b[j]     = Wh[(size_t)(kb + koff + j) * DOUT + n];
            b[j + 8] = Wh[(size_t)(kb + koff + 16 + j) * DOUT + n];
        }
        acc = __builtin_amdgcn_wmma_f32_16x16x32_f16(
            /*neg_a=*/false, a, /*neg_b=*/false, b,
            /*c_mod=*/(short)0, acc, /*reuse_a=*/false, /*reuse_b=*/false);
    }

    const float bn = bias[n];
#pragma unroll
    for (int r = 0; r < 8; ++r) {
        const int m = row0 + hi * 8 + r;
        H[(size_t)m * DOUT + n] = (acc[r] + bn) * inv_s[m];
    }
}

// ------------------------------------------- edge gather + scatter-add (L2)

template <int DOUT>
__global__ void edge_scatter_kernel(const int* __restrict__ snd,
                                    const int* __restrict__ rcv,
                                    const float* __restrict__ H,
                                    float* __restrict__ A) {
    constexpr int CH = DOUT / 4;               // float4 chunks per edge
    const unsigned tid = blockIdx.x * blockDim.x + threadIdx.x;
    if (tid >= (unsigned)N_EDGE * CH) return;
    const int e = tid / CH;
    const int c = tid % CH;
    const int s = snd[e];
    const int r = rcv[e];
    const float4 v = *(const float4*)(H + (size_t)s * DOUT + 4 * c);
    float* dst = A + (size_t)r * DOUT + 4 * c;
    atomicAdd(dst + 0, v.x);
    atomicAdd(dst + 1, v.y);
    atomicAdd(dst + 2, v.z);
    atomicAdd(dst + 3, v.w);
}

// -------------------- fused  x = relu(A * D_r^{-1/2})  ->  f16 for next GEMM

template <int DOUT, bool RELU>
__global__ void finish_to_f16_kernel(const float* __restrict__ A,
                                     const float* __restrict__ inv_r,
                                     _Float16* __restrict__ Xh, int n) {
    int i = blockIdx.x * blockDim.x + threadIdx.x;
    if (i < n) {
        float v = A[i] * inv_r[i / DOUT];
        if (RELU) v = fmaxf(v, 0.0f);
        Xh[i] = (_Float16)v;
    }
}

// ------------------------------- pooling: out[batch[n]] += A[n]*D_r^{-1/2}

__global__ void pool_kernel(const float* __restrict__ A,
                            const float* __restrict__ inv_r,
                            const int* __restrict__ batch,
                            float* __restrict__ out) {
    int i = blockIdx.x * blockDim.x + threadIdx.x;
    if (i < N_NODE * OUT_DIM) {
        const int node = i >> 5;               // OUT_DIM == 32
        const int f    = i & 31;
        atomicAdd(&out[batch[node] * OUT_DIM + f], A[i] * inv_r[node]);
    }
}

// -------------------------------------------------------------------- launch

static inline size_t align256(size_t x) { return (x + 255) & ~(size_t)255; }

extern "C" void kernel_launch(void* const* d_in, const int* in_sizes, int n_in,
                              void* d_out, int out_size, void* d_ws, size_t ws_size,
                              hipStream_t stream) {
    const float* x   = (const float*)d_in[0];
    const float* W1  = (const float*)d_in[1];
    const float* b1  = (const float*)d_in[2];
    const float* W2  = (const float*)d_in[3];
    const float* b2  = (const float*)d_in[4];
    const float* W3  = (const float*)d_in[5];
    const float* b3  = (const float*)d_in[6];
    const int*   snd = (const int*)d_in[7];
    const int*   rcv = (const int*)d_in[8];
    const int*   bat = (const int*)d_in[9];
    float*       out = (float*)d_out;

    // workspace carve-up
    char*  ws    = (char*)d_ws;
    size_t off   = 0;
    float* inv_s = (float*)(ws + off);  off = align256(off + (size_t)N_NODE * 4);
    float* inv_r = (float*)(ws + off);  off = align256(off + (size_t)N_NODE * 4);
    _Float16* Xh = (_Float16*)(ws + off); off = align256(off + (size_t)N_NODE * IN_DIM * 2);
    float* H     = (float*)(ws + off);  off = align256(off + (size_t)N_NODE * HID_DIM * 4);
    float* A     = (float*)(ws + off);  off = align256(off + (size_t)N_NODE * HID_DIM * 4);
    _Float16* Wh = (_Float16*)(ws + off); off = align256(off + (size_t)IN_DIM * HID_DIM * 2);

    const int B = 256;
    auto blocks = [](int n, int b) { return (n + b - 1) / b; };

    // ---- degrees -> D^{-1/2} (shared by all three layers)
    zero_f32_kernel<<<blocks(2 * N_NODE, B), B, 0, stream>>>(inv_s, 2 * N_NODE); // inv_r starts 32 floats past inv_s end (align256 gap)
    degree_accum_kernel<<<blocks(N_EDGE, B), B, 0, stream>>>(snd, rcv, inv_s, inv_r, N_EDGE);
    deg_to_invsqrt_kernel<<<blocks(2 * N_NODE, B), B, 0, stream>>>(inv_s, 2 * N_NODE);

    // ---- layer 1: 128 -> 64, relu
    f32_to_f16_kernel<<<blocks(N_NODE * IN_DIM, B), B, 0, stream>>>(x, Xh, N_NODE * IN_DIM);
    f32_to_f16_kernel<<<blocks(IN_DIM * HID_DIM, B), B, 0, stream>>>(W1, Wh, IN_DIM * HID_DIM);
    gemm_wmma_kernel<IN_DIM, HID_DIM>
        <<<dim3(N_ROW_TILES, HID_DIM / 16), 32, 0, stream>>>(Xh, Wh, b1, inv_s, H);
    zero_f32_kernel<<<blocks(N_NODE * HID_DIM, B), B, 0, stream>>>(A, N_NODE * HID_DIM);
    edge_scatter_kernel<HID_DIM>
        <<<blocks(N_EDGE * (HID_DIM / 4), B), B, 0, stream>>>(snd, rcv, H, A);
    finish_to_f16_kernel<HID_DIM, true>
        <<<blocks(N_NODE * HID_DIM, B), B, 0, stream>>>(A, inv_r, Xh, N_NODE * HID_DIM);

    // ---- layer 2: 64 -> 64, relu
    f32_to_f16_kernel<<<blocks(HID_DIM * HID_DIM, B), B, 0, stream>>>(W2, Wh, HID_DIM * HID_DIM);
    gemm_wmma_kernel<HID_DIM, HID_DIM>
        <<<dim3(N_ROW_TILES, HID_DIM / 16), 32, 0, stream>>>(Xh, Wh, b2, inv_s, H);
    zero_f32_kernel<<<blocks(N_NODE * HID_DIM, B), B, 0, stream>>>(A, N_NODE * HID_DIM);
    edge_scatter_kernel<HID_DIM>
        <<<blocks(N_EDGE * (HID_DIM / 4), B), B, 0, stream>>>(snd, rcv, H, A);
    finish_to_f16_kernel<HID_DIM, true>
        <<<blocks(N_NODE * HID_DIM, B), B, 0, stream>>>(A, inv_r, Xh, N_NODE * HID_DIM);

    // ---- layer 3: 64 -> 32, no relu; pooling fuses the D_r^{-1/2} scale
    f32_to_f16_kernel<<<blocks(HID_DIM * OUT_DIM, B), B, 0, stream>>>(W3, Wh, HID_DIM * OUT_DIM);
    gemm_wmma_kernel<HID_DIM, OUT_DIM>
        <<<dim3(N_ROW_TILES, OUT_DIM / 16), 32, 0, stream>>>(Xh, Wh, b3, inv_s, H);
    zero_f32_kernel<<<blocks(N_NODE * OUT_DIM, B), B, 0, stream>>>(A, N_NODE * OUT_DIM);
    edge_scatter_kernel<OUT_DIM>
        <<<blocks(N_EDGE * (OUT_DIM / 4), B), B, 0, stream>>>(snd, rcv, H, A);

    zero_f32_kernel<<<blocks(NUM_GRAPHS * OUT_DIM, B), B, 0, stream>>>(out, NUM_GRAPHS * OUT_DIM);
    pool_kernel<<<blocks(N_NODE * OUT_DIM, B), B, 0, stream>>>(A, inv_r, bat, out);
}
